// CrossAttention_1743756722493
// MI455X (gfx1250) — compile-verified, matrix operand order checked by reading, not run on hardware
//
#include <hip/hip_runtime.h>

// ---------------------------------------------------------------------------
// CrossAttention for MI455X (gfx1250, wave32, WMMA).
// Pipeline: [Q,K proj] [Vt proj] [S=QK^T] [softmax] [O=P*V], all matmuls via
// v_wmma_f32_16x16x32_bf16 (bf16 inputs, fp32 accumulate).
// bf16-sourced tiles are staged with GLOBAL_LOAD_ASYNC_TO_LDS_B128 (ASYNCcnt),
// fp32-sourced tiles go global->VGPR->convert->LDS.
// ---------------------------------------------------------------------------

typedef __attribute__((ext_vector_type(16))) __bf16 v16bf;
typedef __attribute__((ext_vector_type(8)))  float  v8f;

#define LDS_STRIDE 40   // halves per LDS tile row (80B: 16B-aligned, conflict-free)

union Frag  { v16bf v; uint4 q[2]; };
union Stage { __bf16 b[16]; uint4 q[2]; };

// --- CDNA5 async global->LDS copy (per-lane, 16B), tracked by ASYNCcnt ------
__device__ __forceinline__ unsigned lds_off32(const void* p) {
    // generic LDS pointer: aperture in [63:32], workgroup-relative offset in [31:0]
    return (unsigned)(unsigned long long)p;
}
__device__ __forceinline__ void async_ld_b128(unsigned lds_off, const void* g) {
    asm volatile("global_load_async_to_lds_b128 %0, %1, off"
                 :: "v"(lds_off), "v"(g) : "memory");
}
__device__ __forceinline__ void wait_asynccnt0() {
    asm volatile("s_wait_asynccnt 0x0" ::: "memory");
}

// load 16 contiguous fp32 elements, converting to bf16 halves in registers
__device__ __forceinline__ void ld16(const float* p, Stage& s) {
    const float4* q = (const float4*)p;
#pragma unroll
    for (int i = 0; i < 4; ++i) {
        float4 t = q[i];
        s.b[i*4+0] = (__bf16)t.x; s.b[i*4+1] = (__bf16)t.y;
        s.b[i*4+2] = (__bf16)t.z; s.b[i*4+3] = (__bf16)t.w;
    }
}

__device__ __forceinline__ void st(float* p, float v)          { *p = v; }
__device__ __forceinline__ void st(unsigned short* p, float v) { *(__bf16*)p = (__bf16)v; }

// Fragment fetch from a 128x32 bf16 LDS tile (row-major, LDS_STRIDE halves).
// 16-bit A-matrix 16x32 layout: lanes 0-15 hold K {0..7,16..23},
// lanes 16-31 hold K {8..15,24..31}; B operand uses the symmetric layout
// (lane = column, per-lane K-contiguous) so both load identically.
__device__ __forceinline__ v16bf ldfrag(const unsigned short* base, int row16, int lane) {
    const int row = row16 + (lane & 15);
    const int kb  = (lane < 16) ? 0 : 8;
    const unsigned short* p = base + row * LDS_STRIDE + kb;
    Frag f;
    f.q[0] = *(const uint4*)(p);
    f.q[1] = *(const uint4*)(p + 16);
    return f.v;
}

// Generic GEMM: C[m,n] = sum_k A[m,k] * B[n,k]   (both operands K-contiguous)
// Block tile 128x128, 8 waves of 64x32, K-step 32, double-buffered LDS.
template <typename TA, typename TB, typename TC>
__global__ __launch_bounds__(256)
void gemm_wmma(const TA* __restrict__ Abase, const TB* __restrict__ Bbase,
               TC* __restrict__ Cbase,
               int K, int lda, int ldb, int ldc,
               long sA, long sB, long sC)
{
    constexpr bool ASYNC_A = (sizeof(TA) == 2);   // bf16 source: async path
    constexpr bool ASYNC_B = (sizeof(TB) == 2);

    __shared__ unsigned short As[2][128 * LDS_STRIDE];
    __shared__ unsigned short Bs[2][128 * LDS_STRIDE];

    const int bz = blockIdx.z;
    const TA* A = Abase + bz * sA + (long)(blockIdx.y * 128) * lda;
    const TB* B = Bbase + bz * sB + (long)(blockIdx.x * 128) * ldb;
    TC*       C = Cbase + bz * sC + (long)(blockIdx.y * 128) * ldc + blockIdx.x * 128;

    const int tid  = threadIdx.x;
    const int lane = tid & 31;
    const int wid  = tid >> 5;
    const int wm   = wid >> 2;   // 0..1 : 64 output rows per wave
    const int wn   = wid & 3;    // 0..3 : 32 output cols per wave

    // global->LDS staging assignment: thread covers (row, 16-elem half-row)
    const int grow = tid >> 1;
    const int gcol = (tid & 1) * 16;

    v8f acc[4][2];
#pragma unroll
    for (int mi = 0; mi < 4; ++mi)
#pragma unroll
        for (int ni = 0; ni < 2; ++ni) acc[mi][ni] = (v8f){0,0,0,0,0,0,0,0};

    Stage ta, tb;
    const int nsteps = K >> 5;

    // prologue: stage K-step 0 into buffer 0
    if constexpr (ASYNC_A) {
        const TA* g = A + (long)grow * lda + gcol;
        unsigned  o = lds_off32(&As[0][grow * LDS_STRIDE + gcol]);
        async_ld_b128(o, g);
        async_ld_b128(o + 16, g + 8);
    } else {
        ld16(A + (long)grow * lda + gcol, ta);
        uint4* dA = (uint4*)&As[0][grow * LDS_STRIDE + gcol];
        dA[0] = ta.q[0]; dA[1] = ta.q[1];
    }
    if constexpr (ASYNC_B) {
        const TB* g = B + (long)grow * ldb + gcol;
        unsigned  o = lds_off32(&Bs[0][grow * LDS_STRIDE + gcol]);
        async_ld_b128(o, g);
        async_ld_b128(o + 16, g + 8);
    } else {
        ld16(B + (long)grow * ldb + gcol, tb);
        uint4* dB = (uint4*)&Bs[0][grow * LDS_STRIDE + gcol];
        dB[0] = tb.q[0]; dB[1] = tb.q[1];
    }
    if constexpr (ASYNC_A || ASYNC_B) wait_asynccnt0();
    __syncthreads();

    for (int s = 0; s < nsteps; ++s) {
        const int cur = s & 1, nxt = cur ^ 1;
        const bool more = (s + 1) < nsteps;   // uniform
        if (more) {
            const int k0 = (s + 1) << 5;
            if constexpr (ASYNC_A) {
                const TA* g = A + (long)grow * lda + k0 + gcol;
                unsigned  o = lds_off32(&As[nxt][grow * LDS_STRIDE + gcol]);
                async_ld_b128(o, g);
                async_ld_b128(o + 16, g + 8);
            } else {
                ld16(A + (long)grow * lda + k0 + gcol, ta);
            }
            if constexpr (ASYNC_B) {
                const TB* g = B + (long)grow * ldb + k0 + gcol;
                unsigned  o = lds_off32(&Bs[nxt][grow * LDS_STRIDE + gcol]);
                async_ld_b128(o, g);
                async_ld_b128(o + 16, g + 8);
            } else {
                ld16(B + (long)grow * ldb + k0 + gcol, tb);
            }
        }

        v16bf af[4], bf[2];
#pragma unroll
        for (int mi = 0; mi < 4; ++mi) af[mi] = ldfrag(&As[cur][0], wm * 64 + mi * 16, lane);
#pragma unroll
        for (int ni = 0; ni < 2; ++ni) bf[ni] = ldfrag(&Bs[cur][0], wn * 32 + ni * 16, lane);

#pragma unroll
        for (int mi = 0; mi < 4; ++mi)
#pragma unroll
            for (int ni = 0; ni < 2; ++ni)
                acc[mi][ni] = __builtin_amdgcn_wmma_f32_16x16x32_bf16(
                    false, af[mi], false, bf[ni], (short)0, acc[mi][ni], false, false);

        if (more) {
            if constexpr (!ASYNC_A) {
                uint4* dA = (uint4*)&As[nxt][grow * LDS_STRIDE + gcol];
                dA[0] = ta.q[0]; dA[1] = ta.q[1];
            }
            if constexpr (!ASYNC_B) {
                uint4* dB = (uint4*)&Bs[nxt][grow * LDS_STRIDE + gcol];
                dB[0] = tb.q[0]; dB[1] = tb.q[1];
            }
            if constexpr (ASYNC_A || ASYNC_B) wait_asynccnt0();
        }
        __syncthreads();
    }

    // C/D layout: VGPR r -> rows {r, r+8}; lanes 0-15 / 16-31 -> cols 0-15
    const int row0 = wm * 64 + ((lane >> 4) << 3);
    const int col0 = wn * 32 + (lane & 15);
#pragma unroll
    for (int mi = 0; mi < 4; ++mi)
#pragma unroll
        for (int ni = 0; ni < 2; ++ni)
#pragma unroll
            for (int r = 0; r < 8; ++r)
                st(&C[(long)(row0 + mi * 16 + r) * ldc + col0 + ni * 16], acc[mi][ni][r]);
}

// Row-wise softmax over 2048 fp32 columns, scale folded into exp. In-place.
__global__ __launch_bounds__(256)
void softmax_rows(float* __restrict__ S, float scale)
{
    float* row = S + (long)blockIdx.x * 2048;
    __shared__ float redm[8];
    __shared__ float reds[8];
    const int tid = threadIdx.x;

    float v[8];
    float m = -1e30f;
#pragma unroll
    for (int i = 0; i < 8; ++i) { v[i] = row[tid + i * 256]; m = fmaxf(m, v[i]); }
#pragma unroll
    for (int off = 16; off >= 1; off >>= 1) m = fmaxf(m, __shfl_xor(m, off, 32));
    if ((tid & 31) == 0) redm[tid >> 5] = m;
    __syncthreads();
    float bm = redm[0];
#pragma unroll
    for (int j = 1; j < 8; ++j) bm = fmaxf(bm, redm[j]);

    float sum = 0.f;
#pragma unroll
    for (int i = 0; i < 8; ++i) { v[i] = __expf(scale * (v[i] - bm)); sum += v[i]; }
#pragma unroll
    for (int off = 16; off >= 1; off >>= 1) sum += __shfl_xor(sum, off, 32);
    if ((tid & 31) == 0) reds[tid >> 5] = sum;
    __syncthreads();
    float tot = 0.f;
#pragma unroll
    for (int j = 0; j < 8; ++j) tot += reds[j];

    const float inv = 1.0f / tot;
#pragma unroll
    for (int i = 0; i < 8; ++i) row[tid + i * 256] = v[i] * inv;
}

extern "C" void kernel_launch(void* const* d_in, const int* in_sizes, int n_in,
                              void* d_out, int out_size, void* d_ws, size_t ws_size,
                              hipStream_t stream)
{
    constexpr int B = 8, Sq = 2048, Skv = 2048, D = 1024;
    using ushort_t = unsigned short;

    const float* x   = (const float*)d_in[0];   // [B, Sq, D]
    const float* ctx = (const float*)d_in[1];   // [B, Skv, D]
    const float* Wq  = (const float*)d_in[2];   // [D, D]
    const float* Wk  = (const float*)d_in[3];
    const float* Wv  = (const float*)d_in[4];
    float* out = (float*)d_out;                 // [B, Sq, D]

    // workspace layout
    ushort_t* qb  = (ushort_t*)d_ws;                 // [B*Sq, D]   bf16  32MB
    ushort_t* kb  = qb  + (size_t)B * Sq  * D;       // [B*Skv, D]  bf16  32MB
    ushort_t* vtb = kb  + (size_t)B * Skv * D;       // [B, D, Skv] bf16  32MB
    float*    S   = (float*)(vtb + (size_t)B * D * Skv); // [B, Sq, Skv] 128MB

    const dim3 blk(256);

    // 1) q = x @ Wq^T   (M=B*Sq, N=D, K=D; batch folded into M)
    gemm_wmma<float, float, ushort_t><<<dim3(D/128, (B*Sq)/128, 1), blk, 0, stream>>>(
        x, Wq, qb, D, D, D, D, 0, 0, 0);
    // 2) k = ctx @ Wk^T
    gemm_wmma<float, float, ushort_t><<<dim3(D/128, (B*Skv)/128, 1), blk, 0, stream>>>(
        ctx, Wk, kb, D, D, D, D, 0, 0, 0);
    // 3) vT[b,e,t] = sum_d ctx[b,t,d] * Wv[e,d]   (M=D rows=e, N=Skv cols=t)
    gemm_wmma<float, float, ushort_t><<<dim3(Skv/128, D/128, B), blk, 0, stream>>>(
        Wv, ctx, vtb, D, D, D, Skv,
        0, (long)Skv * D, (long)D * Skv);
    // 4) S[b,s,t] = sum_e q[b,s,e] * k[b,t,e]   (async bf16 tile staging)
    gemm_wmma<ushort_t, ushort_t, float><<<dim3(Skv/128, Sq/128, B), blk, 0, stream>>>(
        qb, kb, S, D, D, D, Skv,
        (long)Sq * D, (long)Skv * D, (long)Sq * Skv);
    // 5) softmax over t with scale = D^-0.5
    softmax_rows<<<dim3(B * Sq), blk, 0, stream>>>(S, 0.03125f);
    // 6) out[b,s,e] = sum_t P[b,s,t] * vT[b,e,t]  (async bf16 B staging)
    gemm_wmma<float, ushort_t, float><<<dim3(D/128, Sq/128, B), blk, 0, stream>>>(
        S, vtb, out, Skv, Skv, Skv, D,
        (long)Sq * Skv, (long)D * Skv, (long)Sq * D);
}
